// DETR_8478265442414
// MI455X (gfx1250) — compile-verified, matrix operand order checked by reading, not run on hardware
//
#include <hip/hip_runtime.h>
#include <hip/hip_bf16.h>

// ---------------- problem constants ----------------
#define N_TOK  1024
#define HDIM   1024
#define DDIM   256
#define QSLOTS 100
#define LSPAN  30
#define WDIM   20
#define SSPAN  30285
#define SPAD   30336   // S padded: multiple of 64 (N-groups/K-steps) and 32 (M)
#define QPAD   128     // Q padded to multiple of 32 (M-tiles of 32)

typedef __attribute__((ext_vector_type(16))) __bf16 v16bf;
typedef __attribute__((ext_vector_type(8)))  __bf16 v8bf;
typedef __attribute__((ext_vector_type(8)))  float  v8f;

// =====================================================================
// bf16 row-major 16x32 tile loader (A, or B^T from [N,K]):
// lane (h*16+r) reads two contiguous 8xbf16 runs [k0+8h,+8) and
// [k0+16+8h,+8) of row r0+r -> two 16B global_load_b128, landing
// directly in WMMA source registers (no cvt in the hot loop).
// Layout per CDNA5 ISA 7.12.2 (16-bit A 16x32). No guards: padded.
// =====================================================================
static __device__ __forceinline__ v16bf load_bf(const __bf16* __restrict__ P, int ld,
                                                int r0, int k0, int lane) {
  const __bf16* p = P + (size_t)(r0 + (lane & 15)) * ld + k0 + 8 * (lane >> 4);
  const v8bf lo = *(const v8bf*)p;
  const v8bf hi = *(const v8bf*)(p + 16);
  return __builtin_shufflevector(lo, hi, 0, 1, 2, 3, 4, 5, 6, 7,
                                 8, 9, 10, 11, 12, 13, 14, 15);
}

// =====================================================================
// Unified bf16 WMMA GEMM:  C = alpha * (A @ Bt^T) + bias
//   A  : [Mp x K] bf16 row-major (Mp%32==0, K%32==0)
//   Bt : [Np x K] bf16 row-major (Np%64==0)
//   C  : mode 0 -> fp32 row-major [Mp x Np]
//        mode 1 -> bf16 row-major [Mp x Np]
//        mode 2 -> bf16 transposed [Np x Mp], one b128 store per tile
// One wave computes a 32x64 block: 2 M-tiles x 4 N-tiles, 8 independent
// WMMA accumulator chains.
// =====================================================================
__global__ void k_gemm(const __bf16* __restrict__ A, const __bf16* __restrict__ Bt,
                       const float* __restrict__ bias, void* __restrict__ Cv,
                       int Mp, int Np, int K, int lda, int ldb, int ldc,
                       float alpha, int mode) {
  const int lane = threadIdx.x & 31;
  const int nG   = Np >> 6;
  const int wave = blockIdx.x * (blockDim.x >> 5) + (threadIdx.x >> 5);
  if (wave >= (Mp >> 5) * nG) return;        // wave-uniform: EXEC stays full
  const int m0 = (wave / nG) << 5;
  const int n0 = (wave % nG) << 6;
  v8f acc[2][4] = {};
  for (int k0 = 0; k0 < K; k0 += 32) {
    if (k0 + 32 < K) {
      __builtin_prefetch(A  + (size_t)(m0 + (lane & 15)) * lda + k0 + 32 + 8 * (lane >> 4), 0, 3);
      __builtin_prefetch(Bt + (size_t)(n0 + (lane & 15)) * ldb + k0 + 32 + 8 * (lane >> 4), 0, 3);
    }
    const v16bf a0 = load_bf(A, lda, m0,      k0, lane);
    const v16bf a1 = load_bf(A, lda, m0 + 16, k0, lane);
    v16bf b[4];
#pragma unroll
    for (int j = 0; j < 4; ++j) b[j] = load_bf(Bt, ldb, n0 + 16 * j, k0, lane);
#pragma unroll
    for (int j = 0; j < 4; ++j) {
      acc[0][j] = __builtin_amdgcn_wmma_f32_16x16x32_bf16(false, a0, false, b[j], (short)0, acc[0][j], false, false);
      acc[1][j] = __builtin_amdgcn_wmma_f32_16x16x32_bf16(false, a1, false, b[j], (short)0, acc[1][j], false, false);
    }
  }
  const int h  = lane >> 4;
  const int nl = lane & 15;
  if (mode == 0) {
    float* C = (float*)Cv;
#pragma unroll
    for (int i = 0; i < 2; ++i)
#pragma unroll
      for (int j = 0; j < 4; ++j) {
        const int n = n0 + 16 * j + nl;
        const float bb = (bias != nullptr) ? bias[n] : 0.f;
#pragma unroll
        for (int r = 0; r < 8; ++r)
          C[(size_t)(m0 + 16 * i + r + 8 * h) * ldc + n] = alpha * acc[i][j][r] + bb;
      }
  } else if (mode == 1) {
    __bf16* C = (__bf16*)Cv;
#pragma unroll
    for (int i = 0; i < 2; ++i)
#pragma unroll
      for (int j = 0; j < 4; ++j) {
        const int n = n0 + 16 * j + nl;
        const float bb = (bias != nullptr) ? bias[n] : 0.f;
#pragma unroll
        for (int r = 0; r < 8; ++r)
          C[(size_t)(m0 + 16 * i + r + 8 * h) * ldc + n] = (__bf16)(alpha * acc[i][j][r] + bb);
      }
  } else {
    // transposed store: lane owns 8 contiguous columns -> one b128 store/tile
    __bf16* C = (__bf16*)Cv;
#pragma unroll
    for (int i = 0; i < 2; ++i)
#pragma unroll
      for (int j = 0; j < 4; ++j) {
        const int n = n0 + 16 * j + nl;
        const float bb = (bias != nullptr) ? bias[n] : 0.f;
        v8bf o;
#pragma unroll
        for (int r = 0; r < 8; ++r) o[r] = (__bf16)(alpha * acc[i][j][r] + bb);
        *(v8bf*)(C + (size_t)n * ldc + m0 + 16 * i + 8 * h) = o;
      }
  }
}

// fp32 -> bf16 elementwise
__global__ void k_cvt(const float* __restrict__ src, __bf16* __restrict__ dst, int n) {
  int i = blockIdx.x * blockDim.x + threadIdx.x;
  if (i < n) dst[i] = (__bf16)src[i];
}

// W[K x N] fp32 -> T[N x K] bf16
__global__ void k_transpose_bf(const float* __restrict__ W, __bf16* __restrict__ T, int K, int N) {
  int idx = blockIdx.x * blockDim.x + threadIdx.x;
  if (idx >= K * N) return;
  int k = idx / N, n = idx - k * N;
  T[(size_t)n * K + k] = (__bf16)W[idx];
}

// =====================================================================
// word attention + stable exp + inclusive prefix sum over the document
// =====================================================================
__global__ void k_wordattn(const float* __restrict__ doc, const float* __restrict__ w,
                           const float* __restrict__ b, float* __restrict__ ebuf,
                           float* __restrict__ prefE) {
  __shared__ float wa[N_TOK];
  __shared__ float sh[N_TOK];
  const int wv = threadIdx.x >> 5, lane = threadIdx.x & 31;
  for (int r = wv; r < N_TOK; r += 32) {     // one wave per row, lanes coalesce
    const float* row = doc + (size_t)r * HDIM;
    float acc = 0.f;
    for (int hh = lane; hh < HDIM; hh += 32) acc += row[hh] * w[hh];
#pragma unroll
    for (int o = 16; o > 0; o >>= 1) acc += __shfl_xor(acc, o, 32);
    if (lane == 0) wa[r] = acc + b[0];
  }
  __syncthreads();
  const int n = threadIdx.x;
  const float score = wa[n];
  sh[n] = score; __syncthreads();
  for (int s = N_TOK / 2; s > 0; s >>= 1) { if (n < s) sh[n] = fmaxf(sh[n], sh[n + s]); __syncthreads(); }
  const float mx = sh[0]; __syncthreads();
  const float ev = expf(score - mx);
  ebuf[n] = ev;
  sh[n] = ev; __syncthreads();
  for (int off = 1; off < N_TOK; off <<= 1) {   // Hillis-Steele inclusive scan
    float t = (n >= off) ? sh[n - off] : 0.f;
    __syncthreads();
    sh[n] += t;
    __syncthreads();
  }
  prefE[n + 1] = sh[n];
  if (n == 0) prefE[0] = 0.f;
}

// widthT[l,d] = span_width_embed[l,:] @ W_width (rows 2H..2H+WD of w_span_proj)
__global__ void k_width(const float* __restrict__ swe, const float* __restrict__ wproj,
                        float* __restrict__ widthT) {
  const int l = blockIdx.x, d = threadIdx.x;
  float acc = 0.f;
  for (int w = 0; w < WDIM; ++w)
    acc += swe[l * WDIM + w] * wproj[(size_t)(2 * HDIM + w) * DDIM + d];
  widthT[(size_t)l * DDIM + d] = acc;
}

// prefEP[t+1,d] = prefEP[t,d] + e[t]*Phead[t,d]
__global__ void k_prefEP(const float* __restrict__ ebuf, const float* __restrict__ Phead,
                         float* __restrict__ prefEP) {
  const int d = threadIdx.x;
  float acc = 0.f;
  prefEP[d] = 0.f;
  for (int t = 0; t < N_TOK; ++t) {
    acc += ebuf[t] * Phead[(size_t)t * DDIM + d];
    prefEP[(size_t)(t + 1) * DDIM + d] = acc;
  }
}

// per-span assembly + input LayerNorm -> fp32 "inputs" output + bf16 ws copy
__global__ void k_span(const float* __restrict__ Pstart, const float* __restrict__ Pend,
                       const float* __restrict__ widthT, const float* __restrict__ prefE,
                       const float* __restrict__ prefEP, const int* __restrict__ starts,
                       const int* __restrict__ ends, const float* __restrict__ bproj,
                       const float* __restrict__ g, const float* __restrict__ beta,
                       float* __restrict__ inputs, __bf16* __restrict__ inputs_bf) {
  __shared__ float red[DDIM];
  const int s = blockIdx.x, d = threadIdx.x;
  const int st = starts[s], en = ends[s];
  int widx = 1 + en - st; if (widx > LSPAN) widx = LSPAN; widx -= 1;
  const float denom = prefE[en + 1] - prefE[st];
  const float head = (prefEP[(size_t)(en + 1) * DDIM + d] - prefEP[(size_t)st * DDIM + d]) / denom;
  const float xd = Pstart[(size_t)st * DDIM + d] + Pend[(size_t)en * DDIM + d]
                 + widthT[(size_t)widx * DDIM + d] + head + bproj[d];
  red[d] = xd; __syncthreads();
  for (int t = DDIM / 2; t > 0; t >>= 1) { if (d < t) red[d] += red[d + t]; __syncthreads(); }
  const float mean = red[0] / DDIM; __syncthreads();
  const float dv = xd - mean;
  red[d] = dv * dv; __syncthreads();
  for (int t = DDIM / 2; t > 0; t >>= 1) { if (d < t) red[d] += red[d + t]; __syncthreads(); }
  const float var = red[0] / DDIM;
  const float outv = dv * rsqrtf(var + 1e-5f) * g[d] + beta[d];
  inputs[(size_t)s * DDIM + d]    = outv;
  inputs_bf[(size_t)s * DDIM + d] = (__bf16)outv;
}

__global__ void k_copy(const float* __restrict__ src, float* __restrict__ dst, int n) {
  int i = blockIdx.x * blockDim.x + threadIdx.x;
  if (i < n) dst[i] = src[i];
}

__global__ void k_zero(float* __restrict__ dst, size_t n) {
  size_t i = (size_t)blockIdx.x * blockDim.x + threadIdx.x;
  if (i < n) dst[i] = 0.f;
}

// q = LN(slots) @ wq + bq -> bf16, one block per slot row
__global__ void k_qproj(const float* __restrict__ slots, const float* __restrict__ g,
                        const float* __restrict__ beta, const float* __restrict__ wq,
                        const float* __restrict__ bq, __bf16* __restrict__ qout) {
  __shared__ float red[DDIM];
  __shared__ float ln[DDIM];
  const int i = blockIdx.x, c = threadIdx.x;
  const float sl = slots[(size_t)i * DDIM + c];
  red[c] = sl; __syncthreads();
  for (int t = DDIM / 2; t > 0; t >>= 1) { if (c < t) red[c] += red[c + t]; __syncthreads(); }
  const float mean = red[0] / DDIM; __syncthreads();
  const float dv = sl - mean;
  red[c] = dv * dv; __syncthreads();
  for (int t = DDIM / 2; t > 0; t >>= 1) { if (c < t) red[c] += red[c + t]; __syncthreads(); }
  const float var = red[0] / DDIM; __syncthreads();
  ln[c] = dv * rsqrtf(var + 1e-5f) * g[c] + beta[c];
  __syncthreads();
  float acc = bq[c];
  for (int d = 0; d < DDIM; ++d) acc += ln[d] * wq[(size_t)d * DDIM + c];
  qout[(size_t)i * DDIM + c] = (__bf16)acc;
}

// softmax over slots (axis=Q) per span column (fp32, in place for attn mode).
// final=1: out = logits [QSLOTS x SSPAN], min(p+eps,1).
__global__ void k_colsm(const float* __restrict__ dots, float* __restrict__ out, int final_) {
  const int j = blockIdx.x * blockDim.x + threadIdx.x;
  if (j >= SSPAN) return;
  float mx = -3.0e38f;
  for (int i = 0; i < QSLOTS; ++i) mx = fmaxf(mx, dots[(size_t)i * SPAD + j]);
  float sum = 0.f;
  for (int i = 0; i < QSLOTS; ++i) sum += expf(dots[(size_t)i * SPAD + j] - mx);
  const float inv = 1.f / sum;
  for (int i = 0; i < QSLOTS; ++i) {
    float a = expf(dots[(size_t)i * SPAD + j] - mx) * inv + 1e-8f;
    if (final_) out[(size_t)i * SSPAN + j] = fminf(a, 1.0f);
    else        out[(size_t)i * SPAD  + j] = a;
  }
}

__global__ void k_rowsum(const float* __restrict__ attn, float* __restrict__ rowsum) {
  __shared__ float red[256];
  const int i = blockIdx.x, c = threadIdx.x;
  float acc = 0.f;
  for (int j = c; j < SSPAN; j += 256) acc += attn[(size_t)i * SPAD + j];
  red[c] = acc; __syncthreads();
  for (int t = 128; t > 0; t >>= 1) { if (c < t) red[c] += red[c + t]; __syncthreads(); }
  if (c == 0) rowsum[i] = red[0];
}

// attn_bf[i,j] = attn[i,j] / rowsum[i]  (rows < QSLOTS, j < SSPAN)
__global__ void k_rownorm(const float* __restrict__ attn, const float* __restrict__ rowsum,
                          __bf16* __restrict__ attn_bf) {
  const int j = blockIdx.x * blockDim.x + threadIdx.x;
  const int i = blockIdx.y;
  if (j < SSPAN) attn_bf[(size_t)i * SPAD + j] = (__bf16)(attn[(size_t)i * SPAD + j] / rowsum[i]);
}

// torch GRUCell on [Q, D]; block = 768 threads, one block per slot row
__global__ void k_gru(const float* __restrict__ upd, float* __restrict__ slots,
                      const float* __restrict__ w_ih, const float* __restrict__ w_hh,
                      const float* __restrict__ b_ih, const float* __restrict__ b_hh) {
  __shared__ float u[DDIM], hh[DDIM], gi[3 * DDIM], gh[3 * DDIM];
  const int i = blockIdx.x, c = threadIdx.x;
  if (c < DDIM) { u[c] = upd[(size_t)i * DDIM + c]; hh[c] = slots[(size_t)i * DDIM + c]; }
  __syncthreads();
  float ai = b_ih[c], ah = b_hh[c];
  for (int d = 0; d < DDIM; ++d) {
    ai += u[d]  * w_ih[(size_t)c * DDIM + d];
    ah += hh[d] * w_hh[(size_t)c * DDIM + d];
  }
  gi[c] = ai; gh[c] = ah;
  __syncthreads();
  if (c < DDIM) {
    const float rg = 1.f / (1.f + expf(-(gi[c] + gh[c])));
    const float zg = 1.f / (1.f + expf(-(gi[DDIM + c] + gh[DDIM + c])));
    const float ng = tanhf(gi[2 * DDIM + c] + rg * gh[2 * DDIM + c]);
    slots[(size_t)i * DDIM + c] = (1.f - zg) * ng + zg * hh[c];
  }
}

// slots += relu(LN(slots)@w1+b1)@w2+b2 ; block = 512 threads, one per slot row
__global__ void k_ff(float* __restrict__ slots, const float* __restrict__ g,
                     const float* __restrict__ beta, const float* __restrict__ w1,
                     const float* __restrict__ b1, const float* __restrict__ w2,
                     const float* __restrict__ b2) {
  __shared__ float red[512];
  __shared__ float ffv[DDIM];
  __shared__ float mid[2 * DDIM];
  const int i = blockIdx.x, c = threadIdx.x;
  float sl = 0.f;
  if (c < DDIM) sl = slots[(size_t)i * DDIM + c];
  red[c] = (c < DDIM) ? sl : 0.f; __syncthreads();
  for (int t = 256; t > 0; t >>= 1) { if (c < t) red[c] += red[c + t]; __syncthreads(); }
  const float mean = red[0] / DDIM; __syncthreads();
  const float dv = sl - mean;
  red[c] = (c < DDIM) ? dv * dv : 0.f; __syncthreads();
  for (int t = 256; t > 0; t >>= 1) { if (c < t) red[c] += red[c + t]; __syncthreads(); }
  const float var = red[0] / DDIM; __syncthreads();
  if (c < DDIM) ffv[c] = dv * rsqrtf(var + 1e-5f) * g[c] + beta[c];
  __syncthreads();
  float m = b1[c];
  for (int d = 0; d < DDIM; ++d) m += ffv[d] * w1[(size_t)d * (2 * DDIM) + c];
  mid[c] = fmaxf(m, 0.f);
  __syncthreads();
  if (c < DDIM) {
    float o = sl + b2[c];
    for (int mm = 0; mm < 2 * DDIM; ++mm) o += mid[mm] * w2[(size_t)mm * DDIM + c];
    slots[(size_t)i * DDIM + c] = o;
  }
}

// =====================================================================
extern "C" void kernel_launch(void* const* d_in, const int* in_sizes, int n_in,
                              void* d_out, int out_size, void* d_ws, size_t ws_size,
                              hipStream_t stream) {
  const float* doc     = (const float*)d_in[0];
  const int*   starts  = (const int*)  d_in[1];
  const int*   ends    = (const int*)  d_in[2];
  const float* w_word  = (const float*)d_in[3];
  const float* b_word  = (const float*)d_in[4];
  const float* swe     = (const float*)d_in[5];
  const float* wproj   = (const float*)d_in[6];
  const float* bproj   = (const float*)d_in[7];
  const float* slots_q = (const float*)d_in[8];
  const float* wq      = (const float*)d_in[9];
  const float* bq      = (const float*)d_in[10];
  const float* wk      = (const float*)d_in[11];
  const float* bk      = (const float*)d_in[12];
  const float* wv      = (const float*)d_in[13];
  const float* bv      = (const float*)d_in[14];
  const float* w_ih    = (const float*)d_in[15];
  const float* w_hh    = (const float*)d_in[16];
  const float* b_ih    = (const float*)d_in[17];
  const float* b_hh    = (const float*)d_in[18];
  const float* w_mlp1  = (const float*)d_in[19];
  const float* b_mlp1  = (const float*)d_in[20];
  const float* w_mlp2  = (const float*)d_in[21];
  const float* b_mlp2  = (const float*)d_in[22];
  const float* g_in    = (const float*)d_in[23];
  const float* beta_in = (const float*)d_in[24];
  const float* g_sl    = (const float*)d_in[25];
  const float* beta_sl = (const float*)d_in[26];
  const float* g_ff    = (const float*)d_in[27];
  const float* beta_ff = (const float*)d_in[28];

  float* out    = (float*)d_out;
  float* logits = out;                                // [Q, S]
  float* inputs = out + (size_t)QSLOTS * SSPAN;       // [S, D]

  // workspace bump allocator (float granularity; bf16 buffers use n/2 floats)
  float* ws = (float*)d_ws;
  size_t off = 0;
  auto alloc   = [&](size_t n) { float* p = ws + off; off += (n + 63) & ~(size_t)63; return p; };
  auto allocbf = [&](size_t n) { return (__bf16*)alloc(n / 2); };   // n even
  float*  ebuf   = alloc(N_TOK);
  float*  prefE  = alloc(N_TOK + 1);
  float*  widthT = alloc((size_t)LSPAN * DDIM);
  __bf16* docbf  = allocbf((size_t)N_TOK * HDIM);
  __bf16* wpT_s  = allocbf((size_t)DDIM * HDIM);      // (w_span_proj start)^T bf16
  __bf16* wpT_e  = allocbf((size_t)DDIM * HDIM);
  __bf16* wpT_h  = allocbf((size_t)DDIM * HDIM);
  __bf16* wkT    = allocbf((size_t)DDIM * DDIM);
  __bf16* wvT    = allocbf((size_t)DDIM * DDIM);
  float*  Pstart = alloc((size_t)N_TOK * DDIM);
  float*  Pend   = alloc((size_t)N_TOK * DDIM);
  float*  Phead  = alloc((size_t)N_TOK * DDIM);
  float*  prefEP = alloc((size_t)(N_TOK + 1) * DDIM);
  __bf16* inbf   = allocbf((size_t)SPAD * DDIM);      // padded bf16 inputs
  __bf16* kbf    = allocbf((size_t)SPAD * DDIM);      // k   [SPAD, D] bf16
  __bf16* vTbf   = allocbf((size_t)DDIM * SPAD);      // v^T [D, SPAD] bf16
  float*  dots   = alloc((size_t)QPAD * SPAD);        // fp32 dots / attn
  __bf16* attnbf = allocbf((size_t)QPAD * SPAD);      // normalized attn bf16
  float*  rowsum = alloc(QSLOTS);
  float*  slots  = alloc((size_t)QSLOTS * DDIM);
  __bf16* qbf    = allocbf((size_t)QPAD * DDIM);
  float*  upd    = alloc((size_t)QPAD * DDIM);

  auto gemm_blocks = [](int Mp, int Np) {
    int waves = (Mp / 32) * (Np / 64);
    return (waves + 3) / 4;                           // 4 waves / block of 128
  };
  auto zgrid = [](size_t n) { return (unsigned)((n + 255) / 256); };

  // 0) deterministic zero-fill of padded bf16 regions (as float writes; counts even)
  k_zero<<<zgrid(((size_t)(SPAD - SSPAN) * DDIM) / 2), 256, 0, stream>>>(
      (float*)(inbf + (size_t)SSPAN * DDIM), ((size_t)(SPAD - SSPAN) * DDIM) / 2);
  k_zero<<<zgrid(((size_t)QPAD * DDIM) / 2), 256, 0, stream>>>(
      (float*)qbf, ((size_t)QPAD * DDIM) / 2);
  k_zero<<<zgrid(((size_t)QPAD * SPAD) / 2), 256, 0, stream>>>(
      (float*)attnbf, ((size_t)QPAD * SPAD) / 2);

  // 0b) one-time conversions: doc -> bf16; weights -> transposed bf16 [N,K]
  k_cvt<<<zgrid((size_t)N_TOK * HDIM), 256, 0, stream>>>(doc, docbf, N_TOK * HDIM);
  k_transpose_bf<<<zgrid((size_t)HDIM * DDIM), 256, 0, stream>>>(wproj, wpT_s, HDIM, DDIM);
  k_transpose_bf<<<zgrid((size_t)HDIM * DDIM), 256, 0, stream>>>(wproj + (size_t)HDIM * DDIM, wpT_e, HDIM, DDIM);
  k_transpose_bf<<<zgrid((size_t)HDIM * DDIM), 256, 0, stream>>>(wproj + (size_t)(2 * HDIM + WDIM) * DDIM, wpT_h, HDIM, DDIM);
  k_transpose_bf<<<zgrid((size_t)DDIM * DDIM), 256, 0, stream>>>(wk, wkT, DDIM, DDIM);
  k_transpose_bf<<<zgrid((size_t)DDIM * DDIM), 256, 0, stream>>>(wv, wvT, DDIM, DDIM);

  // 1) word attention scores, stable exp, prefix sums
  k_wordattn<<<1, N_TOK, 0, stream>>>(doc, w_word, b_word, ebuf, prefE);
  // 2) width embedding projection table
  k_width<<<LSPAN, DDIM, 0, stream>>>(swe, wproj, widthT);
  // 3) hoisted doc-level projections (WMMA): doc @ W_{start,end,head}
  k_gemm<<<gemm_blocks(N_TOK, DDIM), 128, 0, stream>>>(
      docbf, wpT_s, nullptr, Pstart, N_TOK, DDIM, HDIM, HDIM, HDIM, DDIM, 1.f, 0);
  k_gemm<<<gemm_blocks(N_TOK, DDIM), 128, 0, stream>>>(
      docbf, wpT_e, nullptr, Pend,   N_TOK, DDIM, HDIM, HDIM, HDIM, DDIM, 1.f, 0);
  k_gemm<<<gemm_blocks(N_TOK, DDIM), 128, 0, stream>>>(
      docbf, wpT_h, nullptr, Phead,  N_TOK, DDIM, HDIM, HDIM, HDIM, DDIM, 1.f, 0);
  // 4) prefix sums of e[t]*Phead[t,:]
  k_prefEP<<<1, DDIM, 0, stream>>>(ebuf, Phead, prefEP);
  // 5) per-span assembly + input LN (fp32 output #2 + bf16 padded copy)
  k_span<<<SSPAN, DDIM, 0, stream>>>(Pstart, Pend, widthT, prefE, prefEP,
                                     starts, ends, bproj, g_in, beta_in, inputs, inbf);
  // 6) k projection (bf16 row-major) and v projection (bf16 transposed -> v^T)
  k_gemm<<<gemm_blocks(SPAD, DDIM), 128, 0, stream>>>(
      inbf, wkT, bk, kbf,  SPAD, DDIM, DDIM, DDIM, DDIM, DDIM, 1.f, 1);
  k_gemm<<<gemm_blocks(SPAD, DDIM), 128, 0, stream>>>(
      inbf, wvT, bv, vTbf, SPAD, DDIM, DDIM, DDIM, DDIM, SPAD, 1.f, 2);
  // 7) init slots
  k_copy<<<(QSLOTS * DDIM + 255) / 256, 256, 0, stream>>>(slots_q, slots, QSLOTS * DDIM);

  const float scale = 0.0625f;  // D^-0.5
  for (int it = 0; it < 3; ++it) {
    k_qproj<<<QSLOTS, DDIM, 0, stream>>>(slots, g_sl, beta_sl, wq, bq, qbf);
    // dots = scale * q @ k^T
    k_gemm<<<gemm_blocks(QPAD, SPAD), 128, 0, stream>>>(
        qbf, kbf, nullptr, dots, QPAD, SPAD, DDIM, DDIM, DDIM, SPAD, scale, 0);
    // softmax over slots + eps, row sums, normalize -> bf16 attn
    k_colsm<<<(SSPAN + 255) / 256, 256, 0, stream>>>(dots, dots, 0);
    k_rowsum<<<QSLOTS, 256, 0, stream>>>(dots, rowsum);
    k_rownorm<<<dim3((SSPAN + 255) / 256, QSLOTS), 256, 0, stream>>>(dots, rowsum, attnbf);
    // updates = attn @ v = attn @ (v^T)^T  (K = SPAD; padded attn is zero)
    k_gemm<<<gemm_blocks(QPAD, DDIM), 128, 0, stream>>>(
        attnbf, vTbf, nullptr, upd, QPAD, DDIM, SPAD, SPAD, SPAD, DDIM, 1.f, 0);
    k_gru<<<QSLOTS, 3 * DDIM, 0, stream>>>(upd, slots, w_ih, w_hh, b_ih, b_hh);
    k_ff<<<QSLOTS, 2 * DDIM, 0, stream>>>(slots, g_ff, beta_ff, w_mlp1, b_mlp1, w_mlp2, b_mlp2);
  }
  // final logits
  k_qproj<<<QSLOTS, DDIM, 0, stream>>>(slots, g_sl, beta_sl, wq, bq, qbf);
  k_gemm<<<gemm_blocks(QPAD, SPAD), 128, 0, stream>>>(
      qbf, kbf, nullptr, dots, QPAD, SPAD, DDIM, DDIM, DDIM, SPAD, scale, 0);
  k_colsm<<<(SSPAN + 255) / 256, 256, 0, stream>>>(dots, logits, 1);
}
